// CategorySpecificLinear_53798760349899
// MI455X (gfx1250) — matrix operand order repaired and checked by reading, not compiled, not run on hardware
//
#include <hip/hip_runtime.h>
#include <hip/hip_bf16.h>

typedef __bf16 bf16_t;
typedef __attribute__((ext_vector_type(8)))  __bf16 v8bf;
typedef __attribute__((ext_vector_type(16))) __bf16 v16bf;
typedef __attribute__((ext_vector_type(8)))  float  v8f;
typedef __attribute__((ext_vector_type(4)))  float  f32x4;

#define INPUT_DIM  1024
#define HIDDEN_DIM 4096
#define NUM_B      64
#define SEQ        256

#define BM 128
#define BN 256
#define BK 32
#define LDK 40   // padded LDS row stride in bf16 elements (80B, 16B-aligned)

static __device__ __forceinline__ v16bf cat16(v8bf lo, v8bf hi) {
    return __builtin_shufflevector(lo, hi, 0,1,2,3,4,5,6,7,8,9,10,11,12,13,14,15);
}

__global__ __launch_bounds__(256)
void CategorySpecificLinear_wmma_kernel(const float* __restrict__ x,
                                        const int*   __restrict__ cat_ids,
                                        const float* __restrict__ W,
                                        const float* __restrict__ bias,
                                        float* __restrict__ out)
{
    __shared__ __align__(16) bf16_t Abuf[2][BM][LDK];   // [m][k]
    __shared__ __align__(16) bf16_t Bbuf[2][BN][LDK];   // transposed: [n][k]

    const int tid  = threadIdx.x;
    const int lane = tid & 31;
    const int wv   = tid >> 5;      // 0..7
    const int wm   = wv >> 2;       // 0..1 : 64-row M sub-tile
    const int wn   = wv & 3;        // 0..3 : 64-col N sub-tile
    const int lr   = lane & 15;
    const int sel  = lane >> 4;     // 0/1 : which K half this lane holds

    const int n0    = blockIdx.x * BN;       // hidden-dim tile start
    const int row0  = blockIdx.y * BM;       // row among B*S = 16384
    const int batch = row0 / SEQ;
    const int cat   = cat_ids[batch];

    const float* Wc   = W + (size_t)cat * INPUT_DIM * HIDDEN_DIM;
    const float* xblk = x + (size_t)row0 * INPUT_DIM;

    // ---- staging-thread indices ----
    const int am  = tid >> 1;          // 0..127 : A-tile row
    const int ah  = (tid & 1) * 16;    // 0/16   : K half
    const int bn_ = tid;               // 0..255 : B-tile column (Bt row)

    v8f acc[4][4];
    const v8f zacc = {0.f,0.f,0.f,0.f,0.f,0.f,0.f,0.f};
    #pragma unroll
    for (int i = 0; i < 4; ++i)
        #pragma unroll
        for (int j = 0; j < 4; ++j)
            acc[i][j] = zacc;

    auto load_tile = [&](int s, int kt) {
        const int k0 = kt * BK;
        // ---- A: 16 consecutive fp32 per thread -> bf16, 2x16B LDS stores
        const float* ap = xblk + (size_t)am * INPUT_DIM + k0 + ah;
        f32x4 fa0 = *(const f32x4*)(ap + 0);
        f32x4 fa1 = *(const f32x4*)(ap + 4);
        f32x4 fa2 = *(const f32x4*)(ap + 8);
        f32x4 fa3 = *(const f32x4*)(ap + 12);
        v8bf pa0, pa1;
        #pragma unroll
        for (int u = 0; u < 4; ++u) { pa0[u] = (bf16_t)fa0[u]; pa0[4+u] = (bf16_t)fa1[u]; }
        #pragma unroll
        for (int u = 0; u < 4; ++u) { pa1[u] = (bf16_t)fa2[u]; pa1[4+u] = (bf16_t)fa3[u]; }
        *(v8bf*)&Abuf[s][am][ah]     = pa0;
        *(v8bf*)&Abuf[s][am][ah + 8] = pa1;
        // ---- B: one column (fixed n = tid) of 32 K-strided fp32 per thread;
        //         coalesced across the 256 threads at each k; stored [n][k]
        const float* bp = Wc + (size_t)k0 * HIDDEN_DIM + n0 + bn_;
        v8bf pb[4];
        #pragma unroll
        for (int c = 0; c < 4; ++c)
            #pragma unroll
            for (int u = 0; u < 8; ++u)
                pb[c][u] = (bf16_t)bp[(size_t)(c * 8 + u) * HIDDEN_DIM];
        #pragma unroll
        for (int c = 0; c < 4; ++c)
            *(v8bf*)&Bbuf[s][bn_][c * 8] = pb[c];
        // prefetch W panel two K-steps ahead (gfx1250 global_prefetch path)
        if (k0 + 2 * BK < INPUT_DIM)
            __builtin_prefetch(bp + (size_t)2 * BK * HIDDEN_DIM, 0, 0);
    };

    auto compute_tile = [&](int s) {
        v16bf afr[4];
        #pragma unroll
        for (int i = 0; i < 4; ++i) {
            // A 16x32 bf16 layout: lane holds row M=lane%16;
            // contiguous K {sel*16 .. sel*16+15}
            const bf16_t* rp = &Abuf[s][wm * 64 + i * 16 + lr][0];
            afr[i] = cat16(*(const v8bf*)(rp + sel * 16),
                           *(const v8bf*)(rp + sel * 16 + 8));
        }
        v16bf bfr[4];
        #pragma unroll
        for (int j = 0; j < 4; ++j) {
            // B 32x16 bf16 layout: lane holds col N=lane%16;
            // contiguous K {sel*16 .. sel*16+15}
            const bf16_t* rp = &Bbuf[s][wn * 64 + j * 16 + lr][0];
            bfr[j] = cat16(*(const v8bf*)(rp + sel * 16),
                           *(const v8bf*)(rp + sel * 16 + 8));
        }
        #pragma unroll
        for (int i = 0; i < 4; ++i)
            #pragma unroll
            for (int j = 0; j < 4; ++j)
                acc[i][j] = __builtin_amdgcn_wmma_f32_16x16x32_bf16(
                    /*neg_a=*/false, afr[i], /*neg_b=*/false, bfr[j],
                    /*c_mod=*/(short)0, acc[i][j],
                    /*reuse_a=*/false, /*reuse_b=*/false);
    };

    const int NK = INPUT_DIM / BK;   // 32 K-steps
    load_tile(0, 0);
    __syncthreads();
    int cur = 0;
    for (int kt = 0; kt < NK; ++kt) {
        if (kt + 1 < NK) load_tile(cur ^ 1, kt + 1);
        compute_tile(cur);
        __syncthreads();
        cur ^= 1;
    }

    // ---- epilogue: add per-column bias, store fp32 ----
    // C/D layout: VGPR r, lanes 0-15 -> M=r, N=lane; lanes 16-31 -> M=r+8
    #pragma unroll
    for (int j = 0; j < 4; ++j) {
        const int col = n0 + wn * 64 + j * 16 + lr;
        const float bv = bias[(size_t)cat * HIDDEN_DIM + col];
        #pragma unroll
        for (int i = 0; i < 4; ++i) {
            const int mbase = row0 + wm * 64 + i * 16 + sel * 8;
            #pragma unroll
            for (int r = 0; r < 8; ++r) {
                out[(size_t)(mbase + r) * HIDDEN_DIM + col] = acc[i][j][r] + bv;
            }
        }
    }
}

extern "C" void kernel_launch(void* const* d_in, const int* in_sizes, int n_in,
                              void* d_out, int out_size, void* d_ws, size_t ws_size,
                              hipStream_t stream) {
    const float* x       = (const float*)d_in[0];
    const int*   cat_ids = (const int*)d_in[1];
    const float* W       = (const float*)d_in[2];
    const float* bias    = (const float*)d_in[3];
    float*       out     = (float*)d_out;

    dim3 grid(HIDDEN_DIM / BN, (NUM_B * SEQ) / BM);  // 16 x 128
    dim3 block(256);
    CategorySpecificLinear_wmma_kernel<<<grid, block, 0, stream>>>(x, cat_ids, W, bias, out);
}
